// CrossAttention_20435454394511
// MI455X (gfx1250) — compile-verified, matrix operand order checked by reading, not run on hardware
//
#include <hip/hip_runtime.h>

// ---------------- problem constants (from reference) ----------------
#define BB 4
#define CC 512
#define HH 8
#define DH 64
#define NN 2048          // 8*16*16 tokens
#define BN (BB * NN)     // 8192 flattened rows

typedef unsigned short u16;
typedef __attribute__((ext_vector_type(16))) __bf16          v16bf;
typedef __attribute__((ext_vector_type(8)))  float           v8f;
typedef __attribute__((ext_vector_type(16))) unsigned short  v16u;
typedef __attribute__((ext_vector_type(8)))  unsigned short  v8u;

union Frag { v16u u; v16bf bf; };

// round-to-nearest-even f32 -> bf16 (bit manipulation; avoids convert libcalls)
__device__ __forceinline__ u16 f2bf(float f) {
    union { float f; unsigned u; } v; v.f = f;
    unsigned r = v.u + 0x7FFFu + ((v.u >> 16) & 1u);
    return (u16)(r >> 16);
}

// Build a 16x32 bf16 WMMA operand fragment from two contiguous 8-element
// (16-byte) runs.  Per the CDNA5 16-bit A/B layout, lane L (hi = L>>4) holds
// K = {hi*8..hi*8+7} in elements 0..7 and K = {16+hi*8..16+hi*8+7} in 8..15.
__device__ __forceinline__ Frag load_frag(const u16* p0, const u16* p1) {
    Frag f;
    v8u a = *(const v8u*)p0;
    v8u b = *(const v8u*)p1;
#pragma unroll
    for (int i = 0; i < 8; ++i) { f.u[i] = a[i]; f.u[i + 8] = b[i]; }
    return f;
}

__device__ __forceinline__ v8f wmma_bf16(Frag a, Frag b, v8f c) {
    // (neg_a, A, neg_b, B, c_mod, C, reuse_a, reuse_b)
    return __builtin_amdgcn_wmma_f32_16x16x32_bf16(false, a.bf, false, b.bf,
                                                   (short)0, c, false, false);
}

// ---------------- 1: f32 -> bf16 weight convert ----------------
__global__ void ca_cvt_bf16(const float* __restrict__ src, u16* __restrict__ dst, int n) {
    int i = blockIdx.x * blockDim.x + threadIdx.x;
    if (i < n) dst[i] = f2bf(src[i]);
}

// ---------------- 2: x (B,C,N) f32 -> h (B,N,C) bf16 (tiled transpose) ----------------
__global__ void ca_pack_x(const float* __restrict__ x, u16* __restrict__ h) {
    __shared__ float tile[32][33];
    unsigned b  = blockIdx.z;
    unsigned c0 = blockIdx.y * 32;
    unsigned n0 = blockIdx.x * 32;
    unsigned tx = threadIdx.x, ty = threadIdx.y;
#pragma unroll
    for (unsigned i = 0; i < 4; ++i) {
        unsigned c = c0 + ty + 8 * i;
        tile[ty + 8 * i][tx] = x[((size_t)b * CC + c) * NN + n0 + tx];
    }
    __syncthreads();
#pragma unroll
    for (unsigned i = 0; i < 4; ++i) {
        unsigned n = n0 + ty + 8 * i;
        h[((size_t)b * NN + n) * CC + c0 + tx] = f2bf(tile[tx][ty + 8 * i]);
    }
}

// ---------------- 3: projection GEMM: (BN,C)bf16 @ (C,C)bf16 + bias -> (B,H,N,dh)bf16 ----
// block = 128 threads (4 waves); block tile 128(M) x 64(N); each wave owns two
// 16-row strips so every staged B fragment feeds 2 WMMAs.  K-steps of 32.
__global__ void __launch_bounds__(128) ca_proj(
    const u16* __restrict__ h, const u16* __restrict__ W,
    const float* __restrict__ bias, u16* __restrict__ out) {
    __shared__ __align__(16) u16 Bt[64][40];   // W tile transposed [n][k], padded rows
    unsigned blk = blockIdx.x;
    unsigned tb = blk >> 3;            // 128-row token tile (64 tiles)
    unsigned nb = blk & 7;             // 64-col tile (8 tiles)
    unsigned tid = threadIdx.x;
    unsigned w = tid >> 5, lane = tid & 31, hi = lane >> 4, l16 = lane & 15;
    unsigned n0 = nb * 64;
    const u16* arow0 = h + (size_t)(tb * 128 + w * 16 + l16) * CC;
    const u16* arow1 = arow0 + (size_t)64 * CC;

    v8f acc[2][4] = {};
    for (unsigned k0 = 0; k0 < CC; k0 += 32) {
        __syncthreads();
        if (k0 + 32 < CC)
            __builtin_prefetch(&W[(size_t)(k0 + 32) * CC + n0], 0, 0);
#pragma unroll
        for (unsigned i = 0; i < 2; ++i) {      // stage 32x64 W tile, b128 loads
            unsigned cid = i * 128 + tid;       // 256 8-element chunks
            unsigned kk = cid >> 3, nn8 = (cid & 7) * 8;
            v8u wv = *(const v8u*)&W[(size_t)(k0 + kk) * CC + n0 + nn8];
#pragma unroll
            for (unsigned j = 0; j < 8; ++j) Bt[nn8 + j][kk] = wv[j];
        }
        __syncthreads();
        Frag a0 = load_frag(arow0 + k0 + hi * 8, arow0 + k0 + 16 + hi * 8);
        Frag a1 = load_frag(arow1 + k0 + hi * 8, arow1 + k0 + 16 + hi * 8);
#pragma unroll
        for (unsigned c = 0; c < 4; ++c) {
            const u16* bp = &Bt[c * 16 + l16][0];
            Frag bfr = load_frag(bp + hi * 8, bp + 16 + hi * 8);
            acc[0][c] = wmma_bf16(a0, bfr, acc[0][c]);
            acc[1][c] = wmma_bf16(a1, bfr, acc[1][c]);
        }
    }
    // epilogue: D lane layout -> (B,H,N,dh) bf16 with bias (all power-of-2 unsigned math)
#pragma unroll
    for (unsigned m = 0; m < 2; ++m) {
#pragma unroll
        for (unsigned c = 0; c < 4; ++c) {
            unsigned cg = n0 + c * 16 + l16;
            float bv = bias[cg];
            unsigned hd = cg >> 6, d = cg & 63;
#pragma unroll
            for (unsigned r = 0; r < 8; ++r) {
                unsigned tok = tb * 128 + m * 64 + w * 16 + r + 8 * hi;
                unsigned b2 = tok >> 11, n2 = tok & (NN - 1);
                out[(((size_t)b2 * HH + hd) * NN + n2) * DH + d] = f2bf(acc[m][c][r] + bv);
            }
        }
    }
}

// ---------------- 4: streaming softmax stats over one axis ----------------
// s = (A @ B^T)/8.  Per-lane online (max, sumexp) over this lane's column
// subset; single cross-lane log-sum-exp merge at the end (no shfl in k-loop).
// Launch (A=q,B=k) -> row stats; (A=k,B=q) -> column stats of the scores.
__global__ void __launch_bounds__(128) ca_stats(
    const u16* __restrict__ A, const u16* __restrict__ Bm,
    float* __restrict__ omax, float* __restrict__ osum) {
    const unsigned nt = NN / 64;
    unsigned bh = blockIdx.x / nt;
    unsigned qb = blockIdx.x % nt;
    unsigned tid = threadIdx.x, w = tid >> 5, lane = tid & 31, hi = lane >> 4, l16 = lane & 15;
    unsigned q0 = qb * 64 + w * 16;
    const u16* ab = A + ((size_t)bh * NN + q0 + l16) * DH;
    Frag a0 = load_frag(ab + hi * 8,      ab + 16 + hi * 8);
    Frag a1 = load_frag(ab + 32 + hi * 8, ab + 48 + hi * 8);

    float m[8], s[8];
#pragma unroll
    for (int r = 0; r < 8; ++r) { m[r] = -3.0e38f; s[r] = 0.f; }

    for (unsigned kt = 0; kt < NN; kt += 16) {
        const u16* bb = Bm + ((size_t)bh * NN + kt + l16) * DH;
        Frag b0 = load_frag(bb + hi * 8,      bb + 16 + hi * 8);
        Frag b1 = load_frag(bb + 32 + hi * 8, bb + 48 + hi * 8);
        v8f sc = {};
        sc = wmma_bf16(a0, b0, sc);
        sc = wmma_bf16(a1, b1, sc);
#pragma unroll
        for (int r = 0; r < 8; ++r) {
            float v = sc[r] * 0.125f;           // 1/sqrt(64)
            float mn = fmaxf(m[r], v);
            s[r] = s[r] * __expf(m[r] - mn) + __expf(v - mn);
            m[r] = mn;
        }
    }
    // merge the 16 lanes that share each row (log-sum-exp combine)
#pragma unroll
    for (int r = 0; r < 8; ++r) {
        float mm = m[r], ss = s[r];
        for (int off = 1; off < 16; off <<= 1) {
            float mo = __shfl_xor(mm, off, 16);
            float so = __shfl_xor(ss, off, 16);
            float mn = fmaxf(mm, mo);
            ss = ss * __expf(mm - mn) + so * __expf(mo - mn);
            mm = mn;
        }
        m[r] = mm; s[r] = ss;
    }
    if (l16 == 0) {
#pragma unroll
        for (int r = 0; r < 8; ++r) {
            unsigned q = q0 + r + 8 * hi;
            omax[(size_t)bh * NN + q] = m[r];
            osum[(size_t)bh * NN + q] = s[r];
        }
    }
}

// ---------------- 5: context: ctx[row,d] = sum_j exp(s[row,j]-max[j])/sum[j] * V[j,d] ----
// Score tiles recomputed via WMMA; normalized P staged in LDS as bf16 A-fragments.
__global__ void __launch_bounds__(128) ca_ctx(
    const u16* __restrict__ A, const u16* __restrict__ Bm, const u16* __restrict__ V,
    const float* __restrict__ smax, const float* __restrict__ ssum,
    u16* __restrict__ ctx) {
    __shared__ __align__(16) u16 Vt[64][40];        // V tile transposed [d][j]
    __shared__ __align__(16) u16 Pst[4][16][32];    // per-wave P tile [q][j]
    const unsigned nt = NN / 64;
    unsigned bh = blockIdx.x / nt;
    unsigned qb = blockIdx.x % nt;
    unsigned tid = threadIdx.x, w = tid >> 5, lane = tid & 31, hi = lane >> 4, l16 = lane & 15;
    unsigned q0 = qb * 64 + w * 16;
    const u16* ab = A + ((size_t)bh * NN + q0 + l16) * DH;
    Frag a0 = load_frag(ab + hi * 8,      ab + 16 + hi * 8);
    Frag a1 = load_frag(ab + 32 + hi * 8, ab + 48 + hi * 8);

    v8f acc[4] = {};
    for (unsigned j0 = 0; j0 < NN; j0 += 32) {
        if (j0 + 32 < NN)
            __builtin_prefetch(V + ((size_t)bh * NN + j0 + 32) * DH, 0, 0);
#pragma unroll
        for (unsigned i = 0; i < 2; ++i) {          // stage 32x64 V tile, b128 loads
            unsigned cid = i * 128 + tid;
            unsigned jj = cid >> 3, dd8 = (cid & 7) * 8;
            v8u vv = *(const v8u*)&V[((size_t)bh * NN + j0 + jj) * DH + dd8];
#pragma unroll
            for (unsigned j = 0; j < 8; ++j) Vt[dd8 + j][jj] = vv[j];
        }
        // compute two 16x16 score tiles, normalize per reduced index j, stage P
#pragma unroll
        for (unsigned t = 0; t < 2; ++t) {
            unsigned jc = j0 + t * 16 + l16;        // this lane's column
            const u16* bb = Bm + ((size_t)bh * NN + jc) * DH;
            Frag b0 = load_frag(bb + hi * 8,      bb + 16 + hi * 8);
            Frag b1 = load_frag(bb + 32 + hi * 8, bb + 48 + hi * 8);
            v8f sc = {};
            sc = wmma_bf16(a0, b0, sc);
            sc = wmma_bf16(a1, b1, sc);
            float mx  = smax[(size_t)bh * NN + jc];
            float inv = 1.0f / ssum[(size_t)bh * NN + jc];
#pragma unroll
            for (unsigned r = 0; r < 8; ++r) {
                float p = __expf(sc[r] * 0.125f - mx) * inv;
                Pst[w][r + 8 * hi][t * 16 + l16] = f2bf(p);
            }
        }
        __syncthreads();                            // Vt + Pst visible/ordered
        const u16* pr = &Pst[w][l16][0];
        Frag pa = load_frag(pr + hi * 8, pr + 16 + hi * 8);
#pragma unroll
        for (unsigned dt = 0; dt < 4; ++dt) {
            const u16* vr = &Vt[dt * 16 + l16][0];
            Frag vb = load_frag(vr + hi * 8, vr + 16 + hi * 8);
            acc[dt] = wmma_bf16(pa, vb, acc[dt]);
        }
        __syncthreads();                            // before LDS re-stage
    }
#pragma unroll
    for (unsigned dt = 0; dt < 4; ++dt) {
        unsigned d = dt * 16 + l16;
#pragma unroll
        for (unsigned r = 0; r < 8; ++r) {
            unsigned q = q0 + r + 8 * hi;
            ctx[((size_t)bh * NN + q) * DH + d] = f2bf(acc[dt][r]);
        }
    }
}

// ---------------- 6: output projection + bias + residual, write (B,C,N) f32 ----------------
// Same 128x64 block tile / dual-M-strip scheme as ca_proj.
__global__ void __launch_bounds__(128) ca_oproj(
    const u16* __restrict__ ctx, const u16* __restrict__ W,
    const float* __restrict__ bias, const float* __restrict__ xres,
    float* __restrict__ out) {
    __shared__ __align__(16) u16 Bt[64][40];
    unsigned blk = blockIdx.x;
    unsigned tb = blk >> 3, nb = blk & 7;
    unsigned tid = threadIdx.x;
    unsigned w = tid >> 5, lane = tid & 31, hi = lane >> 4, l16 = lane & 15;
    unsigned n0 = nb * 64;

    v8f acc[2][4] = {};
    for (unsigned k0 = 0; k0 < CC; k0 += 32) {
        __syncthreads();
#pragma unroll
        for (unsigned i = 0; i < 2; ++i) {
            unsigned cid = i * 128 + tid;
            unsigned kk = cid >> 3, nn8 = (cid & 7) * 8;
            v8u wv = *(const v8u*)&W[(size_t)(k0 + kk) * CC + n0 + nn8];
#pragma unroll
            for (unsigned j = 0; j < 8; ++j) Bt[nn8 + j][kk] = wv[j];
        }
        __syncthreads();
        unsigned kr0 = k0 + hi * 8, kr1 = k0 + 16 + hi * 8;
#pragma unroll
        for (unsigned m = 0; m < 2; ++m) {
            // A from ctx (B,H,N,dh): 8-element K-runs never cross the 64-wide head block
            unsigned rowf = tb * 128 + m * 64 + w * 16 + l16;
            unsigned b = rowf >> 11, n = rowf & (NN - 1);
            const u16* p0 = ctx + (((size_t)b * HH + (kr0 >> 6)) * NN + n) * DH + (kr0 & 63);
            const u16* p1 = ctx + (((size_t)b * HH + (kr1 >> 6)) * NN + n) * DH + (kr1 & 63);
            Frag a = load_frag(p0, p1);
#pragma unroll
            for (unsigned c = 0; c < 4; ++c) {
                const u16* bp = &Bt[c * 16 + l16][0];
                Frag bfr = load_frag(bp + hi * 8, bp + 16 + hi * 8);
                acc[m][c] = wmma_bf16(a, bfr, acc[m][c]);
            }
        }
    }
    // epilogue: rows are consecutive tokens -> vectorized f32x4 stores with residual
#pragma unroll
    for (unsigned m = 0; m < 2; ++m) {
        unsigned tok0 = tb * 128 + m * 64 + w * 16 + 8 * hi;
        unsigned b = tok0 >> 11, n = tok0 & (NN - 1);
#pragma unroll
        for (unsigned c = 0; c < 4; ++c) {
            unsigned cg = n0 + c * 16 + l16;
            float bv = bias[cg];
            size_t base = ((size_t)b * CC + cg) * NN + n;
            const float4* xr = (const float4*)(xres + base);
            float4* op = (float4*)(out + base);
            float4 x0 = xr[0], x1 = xr[1];
            float4 r0 = make_float4(acc[m][c][0] + bv + x0.x, acc[m][c][1] + bv + x0.y,
                                    acc[m][c][2] + bv + x0.z, acc[m][c][3] + bv + x0.w);
            float4 r1 = make_float4(acc[m][c][4] + bv + x1.x, acc[m][c][5] + bv + x1.y,
                                    acc[m][c][6] + bv + x1.z, acc[m][c][7] + bv + x1.w);
            op[0] = r0; op[1] = r1;
        }
    }
}

// ---------------- host orchestration ----------------
extern "C" void kernel_launch(void* const* d_in, const int* in_sizes, int n_in,
                              void* d_out, int out_size, void* d_ws, size_t ws_size,
                              hipStream_t stream) {
    (void)in_sizes; (void)n_in; (void)out_size; (void)ws_size;
    const float* x1  = (const float*)d_in[0];
    const float* x2  = (const float*)d_in[1];
    const float* Wq  = (const float*)d_in[2];   const float* bq  = (const float*)d_in[3];
    const float* Wk  = (const float*)d_in[4];   const float* bk  = (const float*)d_in[5];
    const float* Wv1 = (const float*)d_in[6];   const float* bv1 = (const float*)d_in[7];
    const float* Wv2 = (const float*)d_in[8];   const float* bv2 = (const float*)d_in[9];
    const float* Wo1 = (const float*)d_in[10];  const float* bo1 = (const float*)d_in[11];
    const float* Wo2 = (const float*)d_in[12];  const float* bo2 = (const float*)d_in[13];
    float* out = (float*)d_out;

    char* ws = (char*)d_ws;
    size_t off = 0;
    auto alloc = [&](size_t bytes) -> char* {
        char* p = ws + off;
        off += (bytes + 255) & ~(size_t)255;
        return p;
    };
    const size_t tokf = (size_t)BN * CC;            // == B*H*N*DH
    u16* h1   = (u16*)alloc(tokf * 2);
    u16* h2   = (u16*)alloc(tokf * 2);
    u16* wqb  = (u16*)alloc((size_t)CC * CC * 2);
    u16* wkb  = (u16*)alloc((size_t)CC * CC * 2);
    u16* wv1b = (u16*)alloc((size_t)CC * CC * 2);
    u16* wv2b = (u16*)alloc((size_t)CC * CC * 2);
    u16* wo1b = (u16*)alloc((size_t)CC * CC * 2);
    u16* wo2b = (u16*)alloc((size_t)CC * CC * 2);
    u16* qb   = (u16*)alloc(tokf * 2);
    u16* kb   = (u16*)alloc(tokf * 2);
    u16* v1b  = (u16*)alloc(tokf * 2);
    u16* v2b  = (u16*)alloc(tokf * 2);
    float* rmax = (float*)alloc((size_t)BB * HH * NN * 4);
    float* rsum = (float*)alloc((size_t)BB * HH * NN * 4);
    float* cmax = (float*)alloc((size_t)BB * HH * NN * 4);
    float* csum = (float*)alloc((size_t)BB * HH * NN * 4);
    // ctx buffers alias h1/h2 (same size; h1/h2 dead after projections)
    u16* ctx1 = h1;
    u16* ctx2 = h2;

    const int nW = CC * CC;
    ca_cvt_bf16<<<(nW + 255) / 256, 256, 0, stream>>>(Wq,  wqb,  nW);
    ca_cvt_bf16<<<(nW + 255) / 256, 256, 0, stream>>>(Wk,  wkb,  nW);
    ca_cvt_bf16<<<(nW + 255) / 256, 256, 0, stream>>>(Wv1, wv1b, nW);
    ca_cvt_bf16<<<(nW + 255) / 256, 256, 0, stream>>>(Wv2, wv2b, nW);
    ca_cvt_bf16<<<(nW + 255) / 256, 256, 0, stream>>>(Wo1, wo1b, nW);
    ca_cvt_bf16<<<(nW + 255) / 256, 256, 0, stream>>>(Wo2, wo2b, nW);

    dim3 pb(32, 8);
    dim3 pg(NN / 32, CC / 32, BB);
    ca_pack_x<<<pg, pb, 0, stream>>>(x1, h1);
    ca_pack_x<<<pg, pb, 0, stream>>>(x2, h2);

    const int gemmBlocks = (BN / 128) * (CC / 64);  // 512
    ca_proj<<<gemmBlocks, 128, 0, stream>>>(h1, wqb,  bq,  qb);
    ca_proj<<<gemmBlocks, 128, 0, stream>>>(h2, wkb,  bk,  kb);
    ca_proj<<<gemmBlocks, 128, 0, stream>>>(h1, wv1b, bv1, v1b);
    ca_proj<<<gemmBlocks, 128, 0, stream>>>(h2, wv2b, bv2, v2b);

    const int attnBlocks = BB * HH * (NN / 64);     // 1024
    ca_stats<<<attnBlocks, 128, 0, stream>>>(qb, kb, rmax, rsum);  // softmax over k (rows)
    ca_stats<<<attnBlocks, 128, 0, stream>>>(kb, qb, cmax, csum);  // softmax over q (cols)

    ca_ctx<<<attnBlocks, 128, 0, stream>>>(qb, kb, v1b, cmax, csum, ctx1);
    ca_ctx<<<attnBlocks, 128, 0, stream>>>(kb, qb, v2b, rmax, rsum, ctx2);

    ca_oproj<<<gemmBlocks, 128, 0, stream>>>(ctx1, wo1b, bo1, x1, out);
    ca_oproj<<<gemmBlocks, 128, 0, stream>>>(ctx2, wo2b, bo2, x2, out + (size_t)BB * CC * NN);
}
